// MLP_VSA_Layer_63531156242779
// MI455X (gfx1250) — matrix operand order repaired and checked by reading, not compiled, not run on hardware
//
#include <hip/hip_runtime.h>
#include <hip/hip_bf16.h>

typedef __attribute__((ext_vector_type(16))) _Float16 v16h;
typedef __attribute__((ext_vector_type(8)))  float    v8f;

#define NPTS  200000
#define NVOX  10000
#define NEDGE (NVOX * 6)

// ---------------------------------------------------------------- utilities
__global__ void fill_u32_kernel(unsigned* p, unsigned v, long long n) {
  long long i = (long long)blockIdx.x * blockDim.x + threadIdx.x;
  if (i < n) p[i] = v;
}

__global__ void relu_copy_kernel(const float* __restrict__ in, float* __restrict__ out, long long n) {
  long long i = (long long)blockIdx.x * blockDim.x + threadIdx.x;
  if (i < n) out[i] = fmaxf(in[i], 0.f);
}

// ------------------------------------------------------------- WMMA GEMM
// C[n x dout] = act(A[n x din, stride lda] @ W[din x dout] + bias)
// Requirements honored by the caller:
//   * n is a multiple of 16
//   * A rows may be over-read up to K-pad (caller guarantees readable memory)
//   * lda is a multiple of 4 (float4-aligned rows)
// One 16x16 output tile per wave; K in chunks of 32 via v_wmma_f32_16x16x32_f16.
__global__ __launch_bounds__(128) void wmma_gemm_kernel(
    const float* __restrict__ A, int lda,
    const float* __restrict__ Wt, const float* __restrict__ bias,
    float* __restrict__ C, int ldc, int n, int din, int dout, int relu) {
  // B tile stored in *fragment order*: [chunk][lane][e] so each lane reads its
  // 16-element fragment as one contiguous 32B LDS access.
  __shared__ __attribute__((aligned(32))) _Float16 Wl[5 * 32 * 16];  // up to K=160
  int colBase = blockIdx.y * 16;
  int din_pad = (din + 31) & ~31;
  for (int idx = threadIdx.x; idx < din_pad * 16; idx += 128) {
    int k = idx >> 4, col = idx & 15;
    float v = (k < din) ? Wt[(size_t)k * dout + colBase + col] : 0.f;
    int c  = k >> 5;          // k-chunk
    int kp = k & 31;          // k within chunk
    int L  = ((kp >> 3) & 1) * 16 + col;   // target lane
    int e  = (kp & 7) + ((kp >> 4) << 3);  // fragment element
    Wl[(c * 32 + L) * 16 + e] = (_Float16)v;
  }
  __syncthreads();
  int wave = threadIdx.x >> 5;
  int lane = threadIdx.x & 31;
  int rowBase = (blockIdx.x * 4 + wave) * 16;
  if (rowBase >= n) return;
  int lh = lane >> 4, ln = lane & 15;
  const float* Arow = A + (size_t)(rowBase + ln) * lda;   // n%16==0 -> always valid
  v8f acc = {};
  for (int k0 = 0; k0 < din_pad; k0 += 32) {
    // A 16x32 f16 fragment (ISA 7.12.2): k = (e<8?0:16) + 8*(lane>>4) + (e&7)
    const float4* ap0 = (const float4*)(Arow + k0 + 8 * lh);
    const float4* ap1 = (const float4*)(Arow + k0 + 16 + 8 * lh);
    float4 a0 = ap0[0], a1 = ap0[1];
    float4 a2 = ap1[0], a3 = ap1[1];
    v16h a;
    a[0]  = (_Float16)a0.x; a[1]  = (_Float16)a0.y; a[2]  = (_Float16)a0.z; a[3]  = (_Float16)a0.w;
    a[4]  = (_Float16)a1.x; a[5]  = (_Float16)a1.y; a[6]  = (_Float16)a1.z; a[7]  = (_Float16)a1.w;
    a[8]  = (_Float16)a2.x; a[9]  = (_Float16)a2.y; a[10] = (_Float16)a2.z; a[11] = (_Float16)a2.w;
    a[12] = (_Float16)a3.x; a[13] = (_Float16)a3.y; a[14] = (_Float16)a3.z; a[15] = (_Float16)a3.w;
    // B fragment: one contiguous 32B LDS read per lane
    v16h b = *(const v16h*)&Wl[((k0 >> 5) * 32 + lane) * 16];
    acc = __builtin_amdgcn_wmma_f32_16x16x32_f16(false, a, false, b, (short)0, acc, false, false);
  }
  float bb = bias[colBase + ln];
#pragma unroll
  for (int e = 0; e < 8; e++) {
    int r = rowBase + e + 8 * lh;           // C/D: M = e + 8*(lane>>4)
    float v = acc[e] + bb;
    if (relu) v = fmaxf(v, 0.f);
    C[(size_t)r * ldc + colBase + ln] = v;
  }
}

// --------------------------------------------------- BatchNorm batch stats
__global__ void col_reduce_kernel(const float* __restrict__ X, int ld, long long n,
                                  float* __restrict__ mean, float* __restrict__ rstd, float eps) {
  int c = blockIdx.x;
  float s = 0.f, s2 = 0.f;
  for (long long r = threadIdx.x; r < n; r += blockDim.x) {
    float v = X[(size_t)r * ld + c];
    s += v; s2 += v * v;
  }
  __shared__ float sh[256], sh2[256];
  sh[threadIdx.x] = s; sh2[threadIdx.x] = s2;
  __syncthreads();
  for (int off = 128; off > 0; off >>= 1) {
    if ((int)threadIdx.x < off) { sh[threadIdx.x] += sh[threadIdx.x + off]; sh2[threadIdx.x] += sh2[threadIdx.x + off]; }
    __syncthreads();
  }
  if (threadIdx.x == 0) {
    float m = sh[0] / (float)n;
    float var = sh2[0] / (float)n - m * m;
    mean[c] = m; rstd[c] = rsqrtf(var + eps);
  }
}

__global__ void bn_apply_kernel(const float* __restrict__ in, float* __restrict__ out,
                                const float* __restrict__ resid,
                                const float* __restrict__ mean, const float* __restrict__ rstd,
                                const float* __restrict__ g, const float* __restrict__ beta,
                                long long total, int d, int relu_after) {
  long long i = (long long)blockIdx.x * blockDim.x + threadIdx.x;
  if (i >= total) return;
  int c = (int)(i % d);
  float v = (in[i] - mean[c]) * rstd[c] * g[c] + beta[c];
  if (relu_after) v = fmaxf(v, 0.f);
  if (resid) v += resid[i];
  out[i] = v;
}

// ------------------------------------------------------- scatter softmax
__device__ __forceinline__ unsigned fmap_u(float f) {
  unsigned u = __float_as_uint(f);
  return (u & 0x80000000u) ? ~u : (u | 0x80000000u);
}
__device__ __forceinline__ float funmap_u(unsigned u) {
  unsigned b = (u & 0x80000000u) ? (u & 0x7FFFFFFFu) : ~u;
  return __uint_as_float(b);
}

__global__ void score_max_kernel(const float* __restrict__ X, const int* __restrict__ inv,
                                 const float* __restrict__ Ws, const float* __restrict__ bs,
                                 float* __restrict__ score, unsigned* __restrict__ m8, int n) {
  __shared__ float W[16 * 8]; __shared__ float B[8];
  if (threadIdx.x < 128) W[threadIdx.x] = Ws[threadIdx.x];
  if (threadIdx.x < 8)   B[threadIdx.x] = bs[threadIdx.x];
  __syncthreads();
  int i = blockIdx.x * blockDim.x + threadIdx.x;
  if (i >= n) return;
  float x[16];
#pragma unroll
  for (int c = 0; c < 16; c++) x[c] = X[(size_t)i * 16 + c];
  int vox = inv[i];
#pragma unroll
  for (int k = 0; k < 8; k++) {
    float s = B[k];
#pragma unroll
    for (int c = 0; c < 16; c++) s += x[c] * W[c * 8 + k];
    score[(size_t)i * 8 + k] = s;
    atomicMax(&m8[vox * 8 + k], fmap_u(s));
  }
}

__global__ void exp_sum_kernel(float* __restrict__ score, const int* __restrict__ inv,
                               const unsigned* __restrict__ m8, float* __restrict__ z8,
                               int* __restrict__ cnt, int n) {
  int i = blockIdx.x * blockDim.x + threadIdx.x;
  if (i >= n) return;
  int vox = inv[i];
#pragma unroll
  for (int k = 0; k < 8; k++) {
    float m = funmap_u(m8[vox * 8 + k]);
    float e = expf(score[(size_t)i * 8 + k] - m);
    score[(size_t)i * 8 + k] = e;
    atomicAdd(&z8[vox * 8 + k], e);
  }
  atomicAdd(&cnt[vox], 1);
}

__global__ void scatter_feat_kernel(const float* __restrict__ e, const float* __restrict__ z8,
                                    const float* __restrict__ X, const int* __restrict__ inv,
                                    float* __restrict__ XV, int n) {
  int i = blockIdx.x * blockDim.x + threadIdx.x;
  if (i >= n) return;
  int vox = inv[i];
  float x[16];
#pragma unroll
  for (int c = 0; c < 16; c++) x[c] = X[(size_t)i * 16 + c];
#pragma unroll
  for (int k = 0; k < 8; k++) {
    float a = e[(size_t)i * 8 + k] / z8[vox * 8 + k];
#pragma unroll
    for (int dd = 0; dd < 16; dd++)
      atomicAdd(&XV[(size_t)vox * 128 + k * 16 + dd], a * x[dd]);
  }
}

// ----------------------------------------------------------------- kNN(6)
__global__ __launch_bounds__(256) void knn_kernel(const int* __restrict__ coords,
                                                  int* __restrict__ knn, int nvox) {
  __shared__ int tile[256 * 4];
  int i = blockIdx.x * 256 + threadIdx.x;
  int bi = 0; float px = 0.f, py = 0.f, pz = 0.f;
  if (i < nvox) {
    bi = coords[i * 4];
    px = (float)coords[i * 4 + 1];
    py = (float)coords[i * 4 + 2];
    pz = (float)coords[i * 4 + 3];
  }
  float bd[6]; int bx[6];
#pragma unroll
  for (int p = 0; p < 6; p++) { bd[p] = 3.0e38f; bx[p] = -1; }
  for (int t0 = 0; t0 < nvox; t0 += 256) {
    int j = t0 + threadIdx.x;
    if (j < nvox) {
      tile[threadIdx.x * 4 + 0] = coords[j * 4 + 0];
      tile[threadIdx.x * 4 + 1] = coords[j * 4 + 1];
      tile[threadIdx.x * 4 + 2] = coords[j * 4 + 2];
      tile[threadIdx.x * 4 + 3] = coords[j * 4 + 3];
    }
    __syncthreads();
    int lim = min(256, nvox - t0);
    if (i < nvox) {
      for (int t = 0; t < lim; t++) {
        int jj = t0 + t;
        if (jj == i) continue;
        if (tile[t * 4] != bi) continue;
        float dx = px - (float)tile[t * 4 + 1];
        float dy = py - (float)tile[t * 4 + 2];
        float dz = pz - (float)tile[t * 4 + 3];
        float d2 = dx * dx + dy * dy + dz * dz;
        if (d2 < bd[5]) {
          int p = 5;
          while (p > 0 && d2 < bd[p - 1]) { bd[p] = bd[p - 1]; bx[p] = bx[p - 1]; p--; }
          bd[p] = d2; bx[p] = jj;
        }
      }
    }
    __syncthreads();
  }
  if (i < nvox) {
#pragma unroll
    for (int p = 0; p < 6; p++) knn[i * 6 + p] = (bx[p] < 0) ? i : bx[p];
  }
}

// ----------------------------------------------- small GEMM 64->3 (relu)
__global__ void gemm3_kernel(const float* __restrict__ A, const float* __restrict__ Wt,
                             const float* __restrict__ b, float* __restrict__ C, int n) {
  __shared__ float W[64 * 3]; __shared__ float B[3];
  for (int t = threadIdx.x; t < 192; t += blockDim.x) W[t] = Wt[t];
  if (threadIdx.x < 3) B[threadIdx.x] = b[threadIdx.x];
  __syncthreads();
  int i = blockIdx.x * blockDim.x + threadIdx.x;
  if (i >= n) return;
  float a0 = B[0], a1 = B[1], a2 = B[2];
  for (int k = 0; k < 64; k++) {
    float a = A[(size_t)i * 64 + k];
    a0 += a * W[k * 3 + 0];
    a1 += a * W[k * 3 + 1];
    a2 += a * W[k * 3 + 2];
  }
  C[i * 3 + 0] = fmaxf(a0, 0.f);
  C[i * 3 + 1] = fmaxf(a1, 0.f);
  C[i * 3 + 2] = fmaxf(a2, 0.f);
}

// --------------------------------------------------------- edge features
// e_in[e] = [pos_j - pos_i + delta_i (3), h_j (128), pad] with e = i*6+p, j = knn[e]
__global__ void edge_build_kernel(const int* __restrict__ knn, const int* __restrict__ coords,
                                  const float* __restrict__ delta, const float* __restrict__ H,
                                  float* __restrict__ EA, long long total) {
  long long idx = (long long)blockIdx.x * blockDim.x + threadIdx.x;
  if (idx >= total) return;
  int e = (int)(idx / 132);
  int c = (int)(idx % 132);
  int i = e / 6;
  int j = knn[e];
  float v;
  if (c < 3)
    v = (float)(coords[j * 4 + 1 + c] - coords[i * 4 + 1 + c]) + delta[i * 3 + c];
  else if (c < 131)
    v = H[(size_t)j * 128 + (c - 3)];
  else
    v = 0.f;
  EA[idx] = v;
}

// dst = repeat(arange(nvox), 6): contiguous 6-edge groups -> trivial max-aggr
__global__ void seg_max6_kernel(const float* __restrict__ E, float* __restrict__ out, int nvox) {
  int idx = blockIdx.x * blockDim.x + threadIdx.x;
  if (idx >= nvox * 128) return;
  int v = idx >> 7, c = idx & 127;
  const float* base = E + (size_t)v * 6 * 128 + c;
  float m = base[0];
#pragma unroll
  for (int p = 1; p < 6; p++) m = fmaxf(m, base[(size_t)p * 128]);
  out[idx] = m;
}

// --------- BN stats of h[inverse].reshape(-1,16) via per-voxel counts ----
__global__ void norm_reduce_kernel(const float* __restrict__ H, const int* __restrict__ cnt,
                                   int nvox, float* __restrict__ mean, float* __restrict__ rstd,
                                   float eps, float ntot) {
  int c = blockIdx.x;   // 0..15
  float s = 0.f, s2 = 0.f;
  for (int v = threadIdx.x; v < nvox; v += blockDim.x) {
    float w = (float)cnt[v];
    const float* row = H + (size_t)v * 128;
    float a = 0.f, b = 0.f;
#pragma unroll
    for (int k = 0; k < 8; k++) { float x = row[k * 16 + c]; a += x; b += x * x; }
    s += w * a; s2 += w * b;
  }
  __shared__ float sh[256], sh2[256];
  sh[threadIdx.x] = s; sh2[threadIdx.x] = s2;
  __syncthreads();
  for (int off = 128; off > 0; off >>= 1) {
    if ((int)threadIdx.x < off) { sh[threadIdx.x] += sh[threadIdx.x + off]; sh2[threadIdx.x] += sh2[threadIdx.x + off]; }
    __syncthreads();
  }
  if (threadIdx.x == 0) {
    float m = sh[0] / ntot;
    float var = sh2[0] / ntot - m * m;
    mean[c] = m; rstd[c] = rsqrtf(var + eps);
  }
}

// ------------------------- final per-point attention + concat(inp, o) ----
__global__ __launch_bounds__(256) void attn_out_kernel(
    const float* __restrict__ X, const float* __restrict__ inp,
    const int* __restrict__ inv, const float* __restrict__ H,
    const float* __restrict__ meanN, const float* __restrict__ rstdN,
    const float* __restrict__ gN, const float* __restrict__ betaN,
    const float* __restrict__ Wq, const float* __restrict__ Wk,
    const float* __restrict__ Wv, const float* __restrict__ Wo,
    const float* __restrict__ bq, const float* __restrict__ bk,
    const float* __restrict__ bv, const float* __restrict__ bo,
    float* __restrict__ out, int n) {
  __shared__ float sWq[256], sWk[256], sWv[256], sWo[256];
  __shared__ float sbq[16], sbk[16], sbv[16], sbo[16], sm[16], sr[16], sg[16], sb2[16];
  for (int t = threadIdx.x; t < 256; t += blockDim.x) {
    sWq[t] = Wq[t]; sWk[t] = Wk[t]; sWv[t] = Wv[t]; sWo[t] = Wo[t];
  }
  if (threadIdx.x < 16) {
    sbq[threadIdx.x] = bq[threadIdx.x]; sbk[threadIdx.x] = bk[threadIdx.x];
    sbv[threadIdx.x] = bv[threadIdx.x]; sbo[threadIdx.x] = bo[threadIdx.x];
    sm[threadIdx.x] = meanN[threadIdx.x]; sr[threadIdx.x] = rstdN[threadIdx.x];
    sg[threadIdx.x] = gN[threadIdx.x]; sb2[threadIdx.x] = betaN[threadIdx.x];
  }
  __syncthreads();
  int i = blockIdx.x * blockDim.x + threadIdx.x;
  if (i >= n) return;
  float x[16];
#pragma unroll
  for (int c = 0; c < 16; c++) x[c] = X[(size_t)i * 16 + c];
  int vox = inv[i];
  const float* hrow = H + (size_t)vox * 128;
  float q[16];
#pragma unroll
  for (int d = 0; d < 16; d++) q[d] = sbq[d];
#pragma unroll
  for (int c = 0; c < 16; c++) {
    float xc = x[c];
#pragma unroll
    for (int d = 0; d < 16; d++) q[d] += xc * sWq[c * 16 + d];
  }
  float logits[8];
  for (int k = 0; k < 8; k++) {
    float kk[16];
#pragma unroll
    for (int d = 0; d < 16; d++) kk[d] = sbk[d];
#pragma unroll
    for (int c = 0; c < 16; c++) {
      float hs = (hrow[k * 16 + c] - sm[c]) * sr[c] * sg[c] + sb2[c];
#pragma unroll
      for (int d = 0; d < 16; d++) kk[d] += hs * sWk[c * 16 + d];
    }
    float acc = 0.f;
#pragma unroll
    for (int d = 0; d < 16; d++) acc += q[d] * kk[d];
    logits[k] = acc * 0.25f;   // 1/sqrt(16)
  }
  float mx = logits[0];
#pragma unroll
  for (int k = 1; k < 8; k++) mx = fmaxf(mx, logits[k]);
  float aw[8]; float z = 0.f;
#pragma unroll
  for (int k = 0; k < 8; k++) { aw[k] = expf(logits[k] - mx); z += aw[k]; }
  float invz = 1.f / z;
  float op[16];
#pragma unroll
  for (int d = 0; d < 16; d++) op[d] = 0.f;
  for (int k = 0; k < 8; k++) {
    float ak = aw[k] * invz;
    float vv[16];
#pragma unroll
    for (int d = 0; d < 16; d++) vv[d] = sbv[d];
#pragma unroll
    for (int c = 0; c < 16; c++) {
      float hs = (hrow[k * 16 + c] - sm[c]) * sr[c] * sg[c] + sb2[c];
#pragma unroll
      for (int d = 0; d < 16; d++) vv[d] += hs * sWv[c * 16 + d];
    }
#pragma unroll
    for (int d = 0; d < 16; d++) op[d] += ak * vv[d];
  }
  float o[16];
#pragma unroll
  for (int d = 0; d < 16; d++) o[d] = sbo[d];
#pragma unroll
  for (int e = 0; e < 16; e++) {
    float oe = op[e];
#pragma unroll
    for (int d = 0; d < 16; d++) o[d] += oe * sWo[e * 16 + d];
  }
#pragma unroll
  for (int c = 0; c < 16; c++) out[(size_t)i * 32 + c] = inp[(size_t)i * 16 + c];
#pragma unroll
  for (int d = 0; d < 16; d++) out[(size_t)i * 32 + 16 + d] = o[d];
}

// ======================================================================
extern "C" void kernel_launch(void* const* d_in, const int* in_sizes, int n_in,
                              void* d_out, int out_size, void* d_ws, size_t ws_size,
                              hipStream_t stream) {
  const float* inp    = (const float*)d_in[0];
  const int*   inv    = (const int*)d_in[1];
  const int*   coords = (const int*)d_in[2];
  // d_in[3] = bev_shape (unused)
  auto F = [&](int i) { return (const float*)d_in[i]; };
  // params pytree leaf order (dict keys sorted): gnn, mhsa, norm, pre_mlp, score
  // per-layer dicts sorted as {b, beta, g, w}
  const int P      = 4;          // gnn[i] base = P + 24*i: f0,f1,g0,g1,h0,h1 (4 leaves each)
  const int MH     = P + 72;     // bk,bo,bq,bv,wk,wo,wq,wv
  const int NORM   = MH + 8;     // beta,g
  const int PRE    = NORM + 2;   // 3 layers x {b,beta,g,w}
  const int SCOREI = PRE + 12;   // b,beta,g,w

  float* W = (float*)d_ws;
  size_t o = 0;
  auto alloc = [&](size_t cnt) { float* p = W + o; o += cnt; return p; };
  float*    XI  = alloc((size_t)NPTS * 16);     // copy of inp (GEMM-safe over-read)
  float*    X0  = alloc((size_t)NPTS * 16);
  float*    X1  = alloc((size_t)NPTS * 16);
  float*    SC  = alloc((size_t)NPTS * 8);
  unsigned* M8  = (unsigned*)alloc(NVOX * 8);
  float*    Z8  = alloc(NVOX * 8);
  int*      CNT = (int*)alloc(NVOX);
  float*    XV  = alloc((size_t)NVOX * 128);
  int*      KNN = (int*)alloc(NVOX * 6);
  float*    H0  = alloc((size_t)NVOX * 128);
  float*    H1  = alloc((size_t)NVOX * 128);
  float*    H2  = alloc((size_t)NVOX * 128);
  float*    H3  = alloc((size_t)NVOX * 128);
  float*    DLT = alloc(NVOX * 3);
  float*    EA  = alloc((size_t)NEDGE * 132);   // e_in, later reused for f-stage2 out
  float*    EB  = alloc((size_t)NEDGE * 128);
  float*    MEAN = alloc(256);
  float*    RSTD = alloc(256);
  (void)alloc(256);                              // tail pad for GEMM K over-read
  (void)ws_size; (void)in_sizes; (void)n_in; (void)out_size;

  auto gup = [](long long n, int b) { return (unsigned)((n + b - 1) / b); };

  auto gemm = [&](const float* A, int lda, const float* Wt, const float* b,
                  float* C, int ldc, long long n, int din, int dout, int relu) {
    dim3 grid(gup(n, 64), dout / 16);
    wmma_gemm_kernel<<<grid, 128, 0, stream>>>(A, lda, Wt, b, C, ldc, (int)n, din, dout, relu);
  };
  auto bnstats = [&](const float* Xp, int ld, long long n, int d, float eps) {
    col_reduce_kernel<<<d, 256, 0, stream>>>(Xp, ld, n, MEAN, RSTD, eps);
  };
  auto bnapply = [&](const float* in, float* out, const float* resid,
                     const float* g, const float* beta, long long n, int d, int relu) {
    bn_apply_kernel<<<gup(n * d, 256), 256, 0, stream>>>(in, out, resid, MEAN, RSTD, g, beta,
                                                         n * d, d, relu);
  };

  // stage inp into ws so every GEMM A-operand tolerates K over-read
  hipMemcpyAsync(XI, inp, (size_t)NPTS * 16 * sizeof(float), hipMemcpyDeviceToDevice, stream);

  // ---- pre_mlp: (Linear -> BN(1e-3) -> ReLU) x2, then Linear -> BN ----
  const float* xin = XI;
  for (int j = 0; j < 3; j++) {
    int base = PRE + 4 * j;
    gemm(xin, 16, F(base + 3), F(base + 0), X1, 16, NPTS, 16, 16, 0);
    bnstats(X1, 16, NPTS, 16, 1e-3f);
    bnapply(X1, X0, nullptr, F(base + 2), F(base + 1), NPTS, 16, (j < 2) ? 1 : 0);
    xin = X0;
  }

  // ---- scatter softmax + voxel pooling ----
  fill_u32_kernel<<<gup(NVOX * 8, 256), 256, 0, stream>>>(M8, 0u, NVOX * 8);
  fill_u32_kernel<<<gup(NVOX * 8, 256), 256, 0, stream>>>((unsigned*)Z8, 0u, NVOX * 8);
  fill_u32_kernel<<<gup(NVOX, 256), 256, 0, stream>>>((unsigned*)CNT, 0u, NVOX);
  fill_u32_kernel<<<gup((long long)NVOX * 128, 256), 256, 0, stream>>>((unsigned*)XV, 0u, (long long)NVOX * 128);
  score_max_kernel<<<gup(NPTS, 256), 256, 0, stream>>>(X0, inv, F(SCOREI + 3), F(SCOREI + 0), SC, M8, NPTS);
  exp_sum_kernel<<<gup(NPTS, 256), 256, 0, stream>>>(SC, inv, M8, Z8, CNT, NPTS);
  scatter_feat_kernel<<<gup(NPTS, 256), 256, 0, stream>>>(SC, Z8, X0, inv, XV, NPTS);

  // ---- kNN graph + h = relu(x_) ----
  knn_kernel<<<gup(NVOX, 256), 256, 0, stream>>>(coords, KNN, NVOX);
  relu_copy_kernel<<<gup((long long)NVOX * 128, 256), 256, 0, stream>>>(XV, H0, (long long)NVOX * 128);

  // ---- 3 PointGNN conv layers ----
  float* hcur = H0;
  float* halt = H3;
  for (int L = 0; L < 3; L++) {
    int base = P + 24 * L;
    // delta = mymlp_h(h): 128->64 (Lin+ReLU+BN), 64->3 (Lin+ReLU+BN), eps=1e-5
    gemm(hcur, 128, F(base + 19), F(base + 16), H1, 64, NVOX, 128, 64, 1);
    bnstats(H1, 64, NVOX, 64, 1e-5f);
    bnapply(H1, H1, nullptr, F(base + 18), F(base + 17), NVOX, 64, 0);
    gemm3_kernel<<<gup(NVOX, 256), 256, 0, stream>>>(H1, F(base + 23), F(base + 20), DLT, NVOX);
    col_reduce_kernel<<<3, 256, 0, stream>>>(DLT, 3, NVOX, MEAN, RSTD, 1e-5f);
    bnapply(DLT, DLT, nullptr, F(base + 22), F(base + 21), NVOX, 3, 0);
    // edge features e_in (60000 x 131, padded stride 132)
    edge_build_kernel<<<gup((long long)NEDGE * 132, 256), 256, 0, stream>>>(
        KNN, coords, DLT, hcur, EA, (long long)NEDGE * 132);
    // f: 131->128, 128->128
    gemm(EA, 132, F(base + 3), F(base + 0), EB, 128, NEDGE, 131, 128, 1);
    bnstats(EB, 128, NEDGE, 128, 1e-5f);
    bnapply(EB, EB, nullptr, F(base + 2), F(base + 1), NEDGE, 128, 0);
    gemm(EB, 128, F(base + 7), F(base + 4), EA, 128, NEDGE, 128, 128, 1);
    bnstats(EA, 128, NEDGE, 128, 1e-5f);
    bnapply(EA, EA, nullptr, F(base + 6), F(base + 5), NEDGE, 128, 0);
    // max aggregation over the 6 edges per voxel
    seg_max6_kernel<<<gup(NVOX * 128, 256), 256, 0, stream>>>(EA, H1, NVOX);
    // g: 128->128, 128->128, then +residual
    gemm(H1, 128, F(base + 11), F(base + 8), H2, 128, NVOX, 128, 128, 1);
    bnstats(H2, 128, NVOX, 128, 1e-5f);
    bnapply(H2, H2, nullptr, F(base + 10), F(base + 9), NVOX, 128, 0);
    gemm(H2, 128, F(base + 15), F(base + 12), H1, 128, NVOX, 128, 128, 1);
    bnstats(H1, 128, NVOX, 128, 1e-5f);
    bnapply(H1, halt, hcur, F(base + 14), F(base + 13), NVOX, 128, 0);
    float* t = hcur; hcur = halt; halt = t;
  }

  // ---- BN stats of h[inverse].reshape(-1,16) via per-voxel counts ----
  norm_reduce_kernel<<<16, 256, 0, stream>>>(hcur, CNT, NVOX, MEAN, RSTD, 1e-3f,
                                             (float)NPTS * 8.0f);
  // ---- per-point single-head attention + concat ----
  attn_out_kernel<<<gup(NPTS, 256), 256, 0, stream>>>(
      X0, inp, inv, hcur, MEAN, RSTD,
      F(NORM + 1), F(NORM + 0),
      F(MH + 6), F(MH + 4), F(MH + 7), F(MH + 5),   // wq, wk, wv, wo
      F(MH + 2), F(MH + 0), F(MH + 3), F(MH + 1),   // bq, bk, bv, bo
      (float*)d_out, NPTS);
}
